// Attention_35433480192333
// MI455X (gfx1250) — matrix operand order repaired and checked by reading, not compiled, not run on hardware
//
#include <hip/hip_runtime.h>
#include <hip/hip_bf16.h>

// ---------------------------------------------------------------------------
// FAVOR+ attention for MI455X (gfx1250), wave32 + WMMA bf16 + TDM staging.
// Stage 1: qkv = x @ W^T          -> q,k,v  [B,H,N,64] bf16   (WMMA 16x16x32 bf16)
// Stage 2: phi = exp(z*s @ rfs - 0.5|z*s|^2)                  (WMMA + v_exp)
// Stage 3: chunked causal linear attention (chunk=64), 1 WG per (b,h),
//          state S (64x64) in LDS, 4 WMMA-GEMMs per chunk,
//          chunk tiles staged LDS-side via tensor_load_to_lds (TENSORcnt).
// ---------------------------------------------------------------------------

#define Bq   8
#define Nseq 4096
#define Cdim 768
#define Hh   12
#define HDim 64
#define Fdim 2304

typedef __bf16 bf16;
typedef __attribute__((ext_vector_type(16))) __bf16 v16bf;
typedef __attribute__((ext_vector_type(8)))  float  v8f;

#if defined(__AMDGCN__) && __has_builtin(__builtin_amdgcn_tensor_load_to_lds) && \
    __has_builtin(__builtin_amdgcn_s_wait_tensorcnt)
#define HAVE_TDM 1
#else
#define HAVE_TDM 0
#endif

#if HAVE_TDM
typedef unsigned tdm_v4u __attribute__((ext_vector_type(4)));
typedef int      tdm_v8i __attribute__((ext_vector_type(8)));
typedef int      tdm_v4i __attribute__((ext_vector_type(4)));

// Issue a TDM DMA of `nelem` contiguous bf16 elements (one 1 x nelem tile)
// from global memory into LDS.  D# layout per CDNA5 ISA §8.3/§8.4.
// This toolchain's builtin takes 6 args (g0, g1, g2, g3, g4, cpol).
__device__ __forceinline__ void tdm_load_row(const bf16* gsrc, bf16* lds_dst, unsigned nelem) {
    unsigned lds_off = (unsigned)(uintptr_t)lds_dst;            // addr[31:0] = LDS offset
    unsigned long long ga = (unsigned long long)(uintptr_t)gsrc;
    tdm_v4u g0;
    g0[0] = 1u;                                                 // count=1 (valid user D#)
    g0[1] = lds_off;                                            // lds_addr
    g0[2] = (unsigned)(ga & 0xffffffffu);                       // global_addr[31:0]
    g0[3] = (unsigned)((ga >> 32) & 0x01ffffffu) | (2u << 30);  // global_addr[56:32], type=2
    tdm_v8i g1;
    g1[0] = (int)(1u << 16);                                    // wg_mask=0, data_size=1 (2B)
    g1[1] = (int)((nelem & 0xffffu) << 16);                     // tensor_dim0[15:0]
    g1[2] = (int)((nelem >> 16) | (1u << 16));                  // tensor_dim0[31:16], tensor_dim1=1
    g1[3] = (int)((nelem & 0xffffu) << 16);                     // tile_dim0 = nelem
    g1[4] = 1;                                                  // tile_dim1=1, tile_dim2=0
    g1[5] = (int)nelem;                                         // tensor_dim0_stride[31:0]
    g1[6] = 0;                                                  // stride hi, dim1_stride lo
    g1[7] = 0;
    tdm_v4i z4 = {0, 0, 0, 0};
    tdm_v8i z8 = {0, 0, 0, 0, 0, 0, 0, 0};
    __builtin_amdgcn_tensor_load_to_lds(g0, g1, z4, z4, z8, 0);
}
#endif

__device__ __forceinline__ v8f vzero() {
    v8f z;
#pragma unroll
    for (int i = 0; i < 8; ++i) z[i] = 0.f;
    return z;
}

__device__ __forceinline__ v8f wmma_bf16(v16bf a, v16bf b, v8f c) {
    // D = A(16x32) * B(32x16) + C, f32 accumulate
    return __builtin_amdgcn_wmma_f32_16x16x32_bf16(
        /*neg_a=*/false, a, /*neg_b=*/false, b,
        /*c_mod=*/(short)0, c, /*reuse_a=*/false, /*reuse_b=*/false);
}

// A fragment: 16x32 bf16 tile, LDS row-major [row][k], row stride ld.
__device__ __forceinline__ v16bf load_a(const bf16* p, int ld, int r0, int k0) {
    int lane = threadIdx.x & 31;
    int row  = r0 + (lane & 15);
    int kh   = (lane >> 4) << 3;
    const bf16* q = p + row * ld + k0 + kh;
    v16bf a;
#pragma unroll
    for (int i = 0; i < 8; ++i) a[i] = q[i];
#pragma unroll
    for (int i = 0; i < 8; ++i) a[8 + i] = q[16 + i];
    return a;
}

// B fragment: 32x16 bf16 (KxN), from an N-major LDS copy BT[n][k] (stride ldK).
__device__ __forceinline__ v16bf load_bt(const bf16* p, int ldK, int n0, int k0) {
    int lane = threadIdx.x & 31;
    int col  = n0 + (lane & 15);
    int kb   = k0 + ((lane >> 4) << 3);
    const bf16* q = p + col * ldK + kb;
    v16bf b;
#pragma unroll
    for (int i = 0; i < 8; ++i) b[i] = q[i];
#pragma unroll
    for (int i = 0; i < 8; ++i) b[8 + i] = q[16 + i];
    return b;
}

// C fragment load from f32 LDS [row][col] (stride ld).
__device__ __forceinline__ v8f load_c(const float* p, int ld, int r0, int n0) {
    int col = n0 + (threadIdx.x & 15);
    int hi  = (threadIdx.x >> 4) & 1;
    v8f c;
#pragma unroll
    for (int i = 0; i < 8; ++i) c[i] = p[(r0 + hi * 8 + i) * ld + col];
    return c;
}

// ---------------------------------------------------------------------------
// Kernel 1: QKV projection.  out[t,f] = sum_k x[t,k] * W[f,k]
// 64(M) x 64(F) tile per 128-thread block, K-steps of 32.
// ---------------------------------------------------------------------------
__global__ __launch_bounds__(128) void qkv_gemm_kernel(
    const float* __restrict__ x, const float* __restrict__ W,
    bf16* __restrict__ qh, bf16* __restrict__ kh, bf16* __restrict__ vh) {
    __shared__ bf16 As[64 * 32];   // [row_t][k]
    __shared__ bf16 Bs[64 * 32];   // [row_f][k]  (N-major B copy)

    const int f0   = blockIdx.x * 64;
    const int m0   = blockIdx.y * 64;
    const int tid  = threadIdx.x;
    const int wave = tid >> 5;
    const int r0   = wave << 4;

    v8f acc[4];
#pragma unroll
    for (int s = 0; s < 4; ++s) acc[s] = vzero();

    const int lrow = tid >> 1;
    const int lcc  = (tid & 1) << 4;

    for (int k0 = 0; k0 < Cdim; k0 += 32) {
        const float* srcx = x + (size_t)(m0 + lrow) * Cdim + k0 + lcc;
        const float* srcw = W + (size_t)(f0 + lrow) * Cdim + k0 + lcc;
        if (k0 + 32 < Cdim) {                 // global_prefetch_b8 for next K tile
            __builtin_prefetch(srcx + 32, 0, 1);
            __builtin_prefetch(srcw + 32, 0, 1);
        }
        // Issue all 8 x b128 global loads up-front, then convert, then store.
        float fx[16], fw[16];
        {
            const float4* sx = (const float4*)srcx;
            const float4* sw = (const float4*)srcw;
            float4 x0 = sx[0], x1 = sx[1], x2 = sx[2], x3 = sx[3];
            float4 w0 = sw[0], w1 = sw[1], w2 = sw[2], w3 = sw[3];
            *(float4*)&fx[0] = x0; *(float4*)&fx[4] = x1;
            *(float4*)&fx[8] = x2; *(float4*)&fx[12] = x3;
            *(float4*)&fw[0] = w0; *(float4*)&fw[4] = w1;
            *(float4*)&fw[8] = w2; *(float4*)&fw[12] = w3;
        }
        bf16* dsta = As + lrow * 32 + lcc;
        bf16* dstw = Bs + lrow * 32 + lcc;
#pragma unroll
        for (int i = 0; i < 16; ++i) dsta[i] = (bf16)fx[i];
#pragma unroll
        for (int i = 0; i < 16; ++i) dstw[i] = (bf16)fw[i];
        __syncthreads();

        v16bf a = load_a(As, 32, r0, 0);
#pragma unroll
        for (int s = 0; s < 4; ++s) {
            v16bf b = load_bt(Bs, 32, s * 16, 0);
            acc[s] = wmma_bf16(a, b, acc[s]);
        }
        __syncthreads();
    }

    // scatter into [B,H,N,64] bf16 buffers
    const int part = f0 / Cdim;
    const int h    = (f0 % Cdim) / HDim;
    bf16* dstbuf = (part == 0) ? qh : ((part == 1) ? kh : vh);
    const int hi = (tid >> 4) & 1;
    const int cl = tid & 15;
#pragma unroll
    for (int s = 0; s < 4; ++s) {
        int d = s * 16 + cl;
#pragma unroll
        for (int i = 0; i < 8; ++i) {
            int t = m0 + r0 + hi * 8 + i;
            int b = t >> 12;               // /Nseq
            int n = t & (Nseq - 1);
            dstbuf[(((size_t)b * Hh + h) * Nseq + n) * HDim + d] = (bf16)acc[s][i];
        }
    }
}

// ---------------------------------------------------------------------------
// Kernel 2: FAVOR+ positive random features.
// phi[n,m] = exp( sum_l z[n,l]*(scale*rfs[h][l][m]) - 0.5*scale^2*|z[n]|^2 )
// ---------------------------------------------------------------------------
__global__ __launch_bounds__(128) void feat_kernel(
    const bf16* __restrict__ z_g, const float* __restrict__ rfs,
    bf16* __restrict__ phi_g) {
    __shared__ bf16 Zs[64 * 64];    // [t][l]
    __shared__ bf16 RT[64 * 64];    // [m][l]  (transposed, pre-scaled)
    __shared__ float rowsq[64];

    const int n0 = blockIdx.x * 64;
    const int h  = blockIdx.y;
    const int b  = blockIdx.z;
    const int tid = threadIdx.x;
    const float scale = 0.35355339059327373f;     // 64^-0.25

    const size_t base = (((size_t)b * Hh + h) * Nseq + n0) * HDim;
    const float* rh = rfs + (size_t)h * HDim * HDim;

#if HAVE_TDM
    if (tid < 32) {                                  // wave 0: TDM copies z tile
        tdm_load_row(z_g + base, Zs, 64 * 64);
        __builtin_amdgcn_s_wait_tensorcnt(0);
    }
#else
#pragma unroll 4
    for (int i = 0; i < 32; ++i) {
        int e = i * 128 + tid;
        Zs[e] = z_g[base + e];
    }
#endif
#pragma unroll 4
    for (int i = 0; i < 32; ++i) {
        int e = i * 128 + tid;
        int l = e >> 6, m = e & 63;
        RT[m * 64 + l] = (bf16)(rh[e] * scale);
    }
    __syncthreads();

    if (tid < 64) {
        float s = 0.f;
#pragma unroll 8
        for (int m = 0; m < 64; ++m) { float zz = (float)Zs[tid * 64 + m]; s += zz * zz; }
        rowsq[tid] = 0.5f * scale * scale * s;
    }
    __syncthreads();

    const int wave = tid >> 5, r0 = wave << 4;
    v8f acc[4];
#pragma unroll
    for (int s = 0; s < 4; ++s) acc[s] = vzero();
#pragma unroll
    for (int l0 = 0; l0 < 64; l0 += 32) {
        v16bf a = load_a(Zs, 64, r0, l0);
#pragma unroll
        for (int s = 0; s < 4; ++s)
            acc[s] = wmma_bf16(a, load_bt(RT, 64, s * 16, l0), acc[s]);
    }

    const int hi = (tid >> 4) & 1, cl = tid & 15;
#pragma unroll
    for (int s = 0; s < 4; ++s) {
        int m = s * 16 + cl;
#pragma unroll
        for (int i = 0; i < 8; ++i) {
            int r = r0 + hi * 8 + i;
            phi_g[base + (size_t)r * HDim + m] = (bf16)__expf(acc[s][i] - rowsq[r]);
        }
    }
}

// ---------------------------------------------------------------------------
// Kernel 3: chunked causal scan.  One workgroup per (b,h), chunks of 64.
//   num = phiq @ S_prev + tril(phiq @ phik^T) @ V
//   den = phiq . ds_prev + rowsum(tril(P))
//   S  += phik^T @ V ;  ds += colsum(phik)
// Chunk tiles staged into LDS by the Tensor Data Mover (wave 0).
// ---------------------------------------------------------------------------
__global__ __launch_bounds__(128) void scan_kernel(
    const bf16* __restrict__ pq_g, const bf16* __restrict__ pk_g,
    const bf16* __restrict__ v_g, float* __restrict__ out) {
    __shared__ bf16 PQ [64 * 64];   // [t][m]
    __shared__ bf16 PK [64 * 64];   // [t][m]
    __shared__ bf16 VS [64 * 64];   // [t][d]
    __shared__ bf16 PKT[64 * 64];   // [m][t]
    __shared__ bf16 VT [64 * 64];   // [d][t]
    __shared__ bf16 Pb [64 * 64];   // [i][j] masked scores
    __shared__ float Sf [64 * 64];  // [m][d] f32 state (C-side)
    __shared__ bf16 SbT[64 * 64];   // [d][m] bf16 state (B-side)
    __shared__ float dsv[64];
    __shared__ float den[64];

    const int bh = blockIdx.x;
    const int b = bh / Hh, h = bh % Hh;
    const int tid = threadIdx.x;
    const int wave = tid >> 5, r0 = wave << 4;
    const int hi = (tid >> 4) & 1, cl = tid & 15;

    for (int i = tid; i < 64 * 64; i += 128) { Sf[i] = 0.f; SbT[i] = (bf16)0.f; }
    if (tid < 64) dsv[tid] = 0.f;
    __syncthreads();

    const size_t hb = ((size_t)b * Hh + h) * (size_t)Nseq * HDim;

    for (int c = 0; c < Nseq / 64; ++c) {
        const size_t base = hb + (size_t)c * 64 * HDim;

        // ---- stage chunk tiles into LDS ----
#if HAVE_TDM
        if (tid < 32) {                          // wave 0 drives the TDM
            tdm_load_row(pq_g + base, PQ, 64 * 64);
            tdm_load_row(pk_g + base, PK, 64 * 64);
            tdm_load_row(v_g  + base, VS, 64 * 64);
            __builtin_amdgcn_s_wait_tensorcnt(0);
        }
#else
#pragma unroll 4
        for (int i = 0; i < 32; ++i) {
            int e = i * 128 + tid;
            PQ[e] = pq_g[base + e];
            PK[e] = pk_g[base + e];
            VS[e] = v_g [base + e];
        }
#endif
        __syncthreads();
        // transposed copies for A(phik^T) and B(V) operands
#pragma unroll 4
        for (int i = 0; i < 32; ++i) {
            int e = i * 128 + tid;
            int t = e >> 6, m = e & 63;
            PKT[m * 64 + t] = PK[e];
            VT [m * 64 + t] = VS[e];
        }
        __syncthreads();

        // --- num_inter = PQ @ S_prev ; P = PQ @ PK^T ---
        v8f acc[4], pacc[4];
#pragma unroll
        for (int s = 0; s < 4; ++s) { acc[s] = vzero(); pacc[s] = vzero(); }
#pragma unroll
        for (int l0 = 0; l0 < 64; l0 += 32) {
            v16bf a = load_a(PQ, 64, r0, l0);
#pragma unroll
            for (int s = 0; s < 4; ++s) {
                acc [s] = wmma_bf16(a, load_bt(SbT, 64, s * 16, l0), acc [s]);
                pacc[s] = wmma_bf16(a, load_bt(PK,  64, s * 16, l0), pacc[s]);
            }
        }
        // causal mask (j <= i) and stash P as bf16
#pragma unroll
        for (int s = 0; s < 4; ++s) {
            int j = s * 16 + cl;
#pragma unroll
            for (int i = 0; i < 8; ++i) {
                int irow = r0 + hi * 8 + i;
                float pv = (j <= irow) ? pacc[s][i] : 0.f;
                Pb[irow * 64 + j] = (bf16)pv;
            }
        }
        __syncthreads();

        // --- den ---
        if (tid < 64) {
            float s = 1e-16f;
#pragma unroll 8
            for (int j = 0; j < 64; ++j) s += (float)Pb[tid * 64 + j];
#pragma unroll 8
            for (int m = 0; m < 64; ++m) s += (float)PQ[tid * 64 + m] * dsv[m];
            den[tid] = s;
        }
        __syncthreads();

        // --- num_intra: acc += Pb @ V ---
#pragma unroll
        for (int l0 = 0; l0 < 64; l0 += 32) {
            v16bf a = load_a(Pb, 64, r0, l0);
#pragma unroll
            for (int s = 0; s < 4; ++s)
                acc[s] = wmma_bf16(a, load_bt(VT, 64, s * 16, l0), acc[s]);
        }

        // --- output: [B,N,C] f32 ---
#pragma unroll
        for (int s = 0; s < 4; ++s) {
            int d = s * 16 + cl;
#pragma unroll
            for (int i = 0; i < 8; ++i) {
                int irow = r0 + hi * 8 + i;
                int n = c * 64 + irow;
                out[((size_t)b * Nseq + n) * Cdim + h * HDim + d] = acc[s][i] / den[irow];
            }
        }
        __syncthreads();

        // --- state update: S += PK^T @ V ---
        {
            v16bf a0 = load_a(PKT, 64, r0, 0);
            v16bf a1 = load_a(PKT, 64, r0, 32);
#pragma unroll
            for (int s = 0; s < 4; ++s) {
                v8f sc = load_c(Sf, 64, r0, s * 16);
                sc = wmma_bf16(a0, load_bt(VT, 64, s * 16, 0),  sc);
                sc = wmma_bf16(a1, load_bt(VT, 64, s * 16, 32), sc);
                int d = s * 16 + cl;
#pragma unroll
                for (int i = 0; i < 8; ++i) {
                    int m = r0 + hi * 8 + i;
                    Sf [m * 64 + d] = sc[i];
                    SbT[d * 64 + m] = (bf16)sc[i];
                }
            }
        }
        if (tid < 64) {
            float s = dsv[tid];
#pragma unroll 8
            for (int t2 = 0; t2 < 64; ++t2) s += (float)PKT[tid * 64 + t2];
            dsv[tid] = s;
        }
        __syncthreads();
    }
}

// ---------------------------------------------------------------------------
extern "C" void kernel_launch(void* const* d_in, const int* in_sizes, int n_in,
                              void* d_out, int out_size, void* d_ws, size_t ws_size,
                              hipStream_t stream) {
    const float* x   = (const float*)d_in[0];   // [B,N,C]
    const float* W   = (const float*)d_in[1];   // [3C,C]
    const float* rfs = (const float*)d_in[2];   // [H,64,64]
    float* out = (float*)d_out;                 // [B,N,C]

    const size_t per = (size_t)Bq * Hh * Nseq * HDim;   // 25,165,824 elems
    bf16* qh = (bf16*)d_ws;
    bf16* kh = qh + per;
    bf16* vh = kh + per;
    bf16* pq = vh + per;
    bf16* pk = pq + per;    // total ws use: 5*per*2 B ~= 252 MB

    qkv_gemm_kernel<<<dim3(Fdim / 64, (Bq * Nseq) / 64), 128, 0, stream>>>(x, W, qh, kh, vh);
    feat_kernel<<<dim3(Nseq / 64, Hh, Bq), 128, 0, stream>>>(qh, rfs, pq);
    feat_kernel<<<dim3(Nseq / 64, Hh, Bq), 128, 0, stream>>>(kh, rfs, pk);
    scan_kernel<<<dim3(Bq * Hh), 128, 0, stream>>>(pq, pk, vh, out);
}
